// Finformer_86861418594907
// MI455X (gfx1250) — matrix operand order repaired and checked by reading, not compiled, not run on hardware
//
#include <hip/hip_runtime.h>

// ---------------------------------------------------------------------------
// Finformer forward for MI455X (gfx1250, wave32, WMMA).
// All matmuls -> v_wmma_f32_16x16x32_f16 (f16 A/B, f32 accum).
// Dynamic-attention scores are produced directly in d_out and softmaxed
// in place (126MB output tensor written once, read once).
// GRU recurrence: LDS-resident state, 60 steps inside one kernel.
// ---------------------------------------------------------------------------

typedef __attribute__((ext_vector_type(16))) _Float16 v16h;
typedef __attribute__((ext_vector_type(8)))  float    v8f;

static constexpr int cN  = 512;
static constexpr int cT  = 60;
static constexpr int cDF = 6;
static constexpr int cH  = 64;
static constexpr int cNH = 2;
static constexpr int cHD = 32;
static constexpr int cE  = 8192;

__device__ __forceinline__ float leakyf(float v)   { return v >= 0.f ? v : 0.01f * v; }
__device__ __forceinline__ float sigmf(float v)    { return 1.f / (1.f + __expf(-v)); }

// ---------------------------------------------------------------------------
// Generic batched WMMA GEMM:  C[z] = alpha * A[z] @ op(B[z]) + bias (+ C[z])
//   z = b1*nb2 + b2 ; per-batch offsets via (sA1,sA2) etc.
//   transB=1: B stored (Ncol,K) row-major (weights, A@W^T)
//   transB=0: B stored (K,Ncol) row-major
// Fragment layouts: A and B both use contiguous k = 16*(lane/16) + s (wave32).
// ---------------------------------------------------------------------------
__global__ __launch_bounds__(256) void k_gemm(
    const float* __restrict__ A, int lda, long long sA1, long long sA2,
    const float* __restrict__ B, int ldb, long long sB1, long long sB2,
    const float* __restrict__ bias,
    float* __restrict__ C, int ldc, long long sC1, long long sC2,
    int M, int Ncol, int K, int nb2, float alpha, int transB, int accum)
{
  const int lane = threadIdx.x & 31;
  const int wave = threadIdx.x >> 5;
  const int z  = blockIdx.z;
  const int b1 = z / nb2, b2 = z - b1 * nb2;
  const float* Ab = A + (long long)b1 * sA1 + (long long)b2 * sA2;
  const float* Bb = B + (long long)b1 * sB1 + (long long)b2 * sB2;
  float*       Cb = C + (long long)b1 * sC1 + (long long)b2 * sC2;

  const int m0 = blockIdx.y * 32 + (wave >> 2) * 16;
  const int n0 = blockIdx.x * 64 + (wave & 3) * 16;
  if (m0 >= M || n0 >= Ncol) return;          // wave-uniform exit

  const int  arow  = m0 + (lane & 15);
  const int  koffA = (lane >> 4) * 16;        // lanes 16-31 hold K 16..31
  const int  bcol  = n0 + (lane & 15);
  const int  koffB = (lane >> 4) * 16;        // lanes 16-31 hold K 16..31, same order as A
  const bool aok   = arow < M;
  const bool bok   = bcol < Ncol;

  v8f acc = {};
  for (int k0 = 0; k0 < K; k0 += 32) {
    if (aok && (k0 + 32) < K)
      __builtin_prefetch(Ab + (long long)arow * lda + k0 + 32, 0, 1);
    v16h af, bf;
#pragma unroll
    for (int j = 0; j < 8; ++j) {
      int ka = k0 + koffA + j;            // VGPR0-3 of A fragment
      int kb = ka + 8;                    // VGPR4-7 of A fragment
      af[j]     = (aok && ka < K) ? (_Float16)Ab[(long long)arow * lda + ka]
                                  : (_Float16)0.f;
      af[j + 8] = (aok && kb < K) ? (_Float16)Ab[(long long)arow * lda + kb]
                                  : (_Float16)0.f;
    }
#pragma unroll
    for (int j = 0; j < 16; ++j) {
      int k = k0 + koffB + j;
      float bv = 0.f;
      if (bok && k < K)
        bv = transB ? Bb[(long long)bcol * ldb + k]
                    : Bb[(long long)k * ldb + bcol];
      bf[j] = (_Float16)bv;
    }
    acc = __builtin_amdgcn_wmma_f32_16x16x32_f16(false, af, false, bf,
                                                 (short)0, acc, false, false);
  }

  if (bok) {
    const float bv = bias ? bias[bcol] : 0.f;
    const int crow0 = m0 + (lane >> 4) * 8;   // ISA C layout: VGPR r -> M=r(+8)
#pragma unroll
    for (int r = 0; r < 8; ++r) {
      int crow = crow0 + r;
      if (crow < M) {
        long long off = (long long)crow * ldc + bcol;
        float v = acc[r] * alpha + bv;
        if (accum) v += Cb[off];
        Cb[off] = v;
      }
    }
  }
}

// ---------------------------------------------------------------------------
// GRU recurrence (one layer). gi = x@Wih^T + bih precomputed (batched GEMM).
// Block owns 64 batch rows; h state lives in LDS (fp32 + f16 mirror),
// Whh lives in LDS as f16; 60 timesteps looped in-kernel with barriers.
// LDS: 16KB hS + 8KB hH + 24KB wS + 48KB gh = 96KB of 320KB/WGP.
// ---------------------------------------------------------------------------
__global__ __launch_bounds__(256) void k_gru(
    const float* __restrict__ GI,   // (N*T, 192), row = n*T + t
    const float* __restrict__ whh,  // (192, 64)
    const float* __restrict__ bhh,  // (192)
    float* __restrict__ G)          // (N*T, 64)
{
  extern __shared__ char smem[];
  float*    hS = (float*)smem;                                   // 64*64 f32
  _Float16* hH = (_Float16*)(smem + 64 * 64 * 4);                // 64*64 f16
  _Float16* wS = (_Float16*)(smem + 64 * 64 * 4 + 64 * 64 * 2);  // 192*64 f16
  float*    gh = (float*)(smem + 64 * 64 * 4 + 64 * 64 * 2 + 192 * 64 * 2);

  const int tid  = threadIdx.x;
  const int lane = tid & 31;
  const int wave = tid >> 5;
  const int n0   = blockIdx.x * 64;

  for (int i = tid; i < 192 * 64; i += 256) wS[i] = (_Float16)whh[i];
  for (int i = tid; i < 64 * 64;  i += 256) { hS[i] = 0.f; hH[i] = (_Float16)0.f; }
  __syncthreads();

  const int arow  = lane & 15;
  const int koffA = (lane >> 4) * 16;
  const int bcol  = lane & 15;
  const int koffB = (lane >> 4) * 16;

  for (int t = 0; t < cT; ++t) {
    // gh(64x192) = hH(64x64) @ wS^T ; 4x12=48 tiles, 6 per wave
#pragma unroll
    for (int i = 0; i < 6; ++i) {
      int tile = wave + (i << 3);
      int mt = tile / 12, nt = tile - mt * 12;
      int mb = mt * 16, nb = nt * 16;
      v8f acc = {};
#pragma unroll
      for (int k0 = 0; k0 < 64; k0 += 32) {
        v16h af, bf;
#pragma unroll
        for (int j = 0; j < 8; ++j) {
          af[j]     = hH[(mb + arow) * 64 + k0 + koffA + j];
          af[j + 8] = hH[(mb + arow) * 64 + k0 + koffA + 8 + j];
        }
#pragma unroll
        for (int j = 0; j < 16; ++j)
          bf[j] = wS[(nb + bcol) * 64 + k0 + koffB + j];
        acc = __builtin_amdgcn_wmma_f32_16x16x32_f16(false, af, false, bf,
                                                     (short)0, acc, false, false);
      }
      int cr0 = mb + (lane >> 4) * 8;
      int cc  = nb + (lane & 15);
#pragma unroll
      for (int r = 0; r < 8; ++r)
        gh[(cr0 + r) * 192 + cc] = acc[r];
    }
    __syncthreads();

    // fused gate math + state update + stream h_t out
    for (int i = tid; i < 64 * 64; i += 256) {
      int rr = i >> 6, c = i & 63;
      long long row = (long long)(n0 + rr) * cT + t;
      const float* gi = GI + row * 192;
      float ghr = gh[rr * 192 +       c] + bhh[c];
      float ghz = gh[rr * 192 +  64 + c] + bhh[64 + c];
      float ghn = gh[rr * 192 + 128 + c] + bhh[128 + c];
      float r_ = sigmf(gi[c]      + ghr);
      float z_ = sigmf(gi[64 + c] + ghz);
      float nn = tanhf(gi[128 + c] + r_ * ghn);
      float hv = (1.f - z_) * nn + z_ * hS[i];
      hS[i] = hv;
      hH[i] = (_Float16)hv;
      G[row * 64 + c] = hv;
    }
    __syncthreads();
  }
}

// ---------------------------------------------------------------------------
// Elementwise / reduction kernels (wave32-per-row where reductions needed)
// ---------------------------------------------------------------------------
__global__ void k_transpose(const float* __restrict__ in, float* __restrict__ outp)
{ // (N,T,H) -> (T,N,H)
  long long idx = (long long)blockIdx.x * blockDim.x + threadIdx.x;
  const long long total = (long long)cN * cT * cH;
  if (idx >= total) return;
  int c = idx & 63;
  long long row = idx >> 6;
  int t = (int)(row / cN);
  int n = (int)(row - (long long)t * cN);
  outp[idx] = in[((long long)n * cT + t) * cH + c];
}

__global__ __launch_bounds__(256) void k_softmax_meanmask(float* __restrict__ d,
                                                          long long rows, int len)
{ // softmax(where(v >= rowmean, v, -1e10)) in place
  long long row = (long long)blockIdx.x * 8 + (threadIdx.x >> 5);
  if (row >= rows) return;
  const int lane = threadIdx.x & 31;
  float* p = d + row * len;
  float s = 0.f;
  for (int i = lane; i < len; i += 32) s += p[i];
  for (int o = 16; o; o >>= 1) s += __shfl_xor(s, o, 32);
  const float mean = s / (float)len;
  float mx = -3.0e38f;
  for (int i = lane; i < len; i += 32) { float v = p[i]; v = v >= mean ? v : -1.0e10f; mx = fmaxf(mx, v); }
  for (int o = 16; o; o >>= 1) mx = fmaxf(mx, __shfl_xor(mx, o, 32));
  float se = 0.f;
  for (int i = lane; i < len; i += 32) { float v = p[i]; v = v >= mean ? v : -1.0e10f; se += __expf(v - mx); }
  for (int o = 16; o; o >>= 1) se += __shfl_xor(se, o, 32);
  const float inv = 1.f / se;
  for (int i = lane; i < len; i += 32) { float v = p[i]; v = v >= mean ? v : -1.0e10f; p[i] = __expf(v - mx) * inv; }
}

__global__ __launch_bounds__(256) void k_softmax_adjmask(float* __restrict__ d,
    const unsigned char* __restrict__ adj, long long rows)
{ // rows = T*N, len = N; node n = row % N
  long long row = (long long)blockIdx.x * 8 + (threadIdx.x >> 5);
  if (row >= rows) return;
  const int lane = threadIdx.x & 31;
  const int n = (int)(row & (cN - 1));
  float* p = d + row * cN;
  const unsigned char* am = adj + (long long)n * cN;
  float mx = -3.0e38f;
  for (int i = lane; i < cN; i += 32) { float v = am[i] ? p[i] : -1.0e10f; mx = fmaxf(mx, v); }
  for (int o = 16; o; o >>= 1) mx = fmaxf(mx, __shfl_xor(mx, o, 32));
  float se = 0.f;
  for (int i = lane; i < cN; i += 32) { float v = am[i] ? p[i] : -1.0e10f; se += __expf(v - mx); }
  for (int o = 16; o; o >>= 1) se += __shfl_xor(se, o, 32);
  const float inv = 1.f / se;
  for (int i = lane; i < cN; i += 32) { float v = am[i] ? p[i] : -1.0e10f; p[i] = __expf(v - mx) * inv; }
}

__global__ __launch_bounds__(256) void k_softmax(float* __restrict__ d,
                                                 long long rows, int len)
{
  long long row = (long long)blockIdx.x * 8 + (threadIdx.x >> 5);
  if (row >= rows) return;
  const int lane = threadIdx.x & 31;
  float* p = d + row * len;
  float mx = -3.0e38f;
  for (int i = lane; i < len; i += 32) mx = fmaxf(mx, p[i]);
  for (int o = 16; o; o >>= 1) mx = fmaxf(mx, __shfl_xor(mx, o, 32));
  float se = 0.f;
  for (int i = lane; i < len; i += 32) se += __expf(p[i] - mx);
  for (int o = 16; o; o >>= 1) se += __shfl_xor(se, o, 32);
  const float inv = 1.f / se;
  for (int i = lane; i < len; i += 32) p[i] = __expf(p[i] - mx) * inv;
}

__global__ void k_zero8(unsigned char* __restrict__ p, long long n)
{
  long long i = (long long)blockIdx.x * blockDim.x + threadIdx.x;
  if (i < n) p[i] = 0;
}
__global__ void k_adj_scatter(const int* __restrict__ ei, unsigned char* __restrict__ adj)
{
  int e = blockIdx.x * blockDim.x + threadIdx.x;
  if (e < cE) adj[(long long)ei[e] * cN + ei[cE + e]] = 1;
}

__global__ __launch_bounds__(256) void k_res_ln1(
    const float* __restrict__ dyn,  // (T,NH,N,HD)
    const float* __restrict__ stat, // (T*N,H)
    const float* __restrict__ xs,   // (T*N,H)
    const float* __restrict__ g, const float* __restrict__ b,
    float* __restrict__ res, float* __restrict__ outp, long long rows)
{
  long long row = (long long)blockIdx.x * 8 + (threadIdx.x >> 5);
  if (row >= rows) return;
  const int lane = threadIdx.x & 31;
  int t = (int)(row / cN), n = (int)(row - (long long)t * cN);
  float v[2]; float s = 0.f, s2 = 0.f;
#pragma unroll
  for (int j = 0; j < 2; ++j) {
    int c = lane + j * 32;
    int hh = c >> 5, dd = c & 31;
    float dv = dyn[(((long long)t * cNH + hh) * cN + n) * cHD + dd];
    float val = dv + stat[row * cH + c] + xs[row * cH + c];
    v[j] = val; s += val; s2 += val * val;
  }
  for (int o = 16; o; o >>= 1) { s += __shfl_xor(s, o, 32); s2 += __shfl_xor(s2, o, 32); }
  float mean = s / (float)cH;
  float inv  = rsqrtf(s2 / (float)cH - mean * mean + 1e-5f);
#pragma unroll
  for (int j = 0; j < 2; ++j) {
    int c = lane + j * 32;
    res[row * cH + c]  = v[j];
    outp[row * cH + c] = (v[j] - mean) * inv * g[c] + b[c];
  }
}

__global__ __launch_bounds__(256) void k_ln2_pe(
    const float* __restrict__ o2, const float* __restrict__ res,
    const float* __restrict__ g, const float* __restrict__ b,
    float* __restrict__ src, long long rows)
{
  long long row = (long long)blockIdx.x * 8 + (threadIdx.x >> 5);
  if (row >= rows) return;
  const int lane = threadIdx.x & 31;
  const int t = (int)(row / cN);
  float v[2]; float s = 0.f, s2 = 0.f;
#pragma unroll
  for (int j = 0; j < 2; ++j) {
    int c = lane + j * 32;
    float val = o2[row * cH + c] + res[row * cH + c];
    v[j] = val; s += val; s2 += val * val;
  }
  for (int o = 16; o; o >>= 1) { s += __shfl_xor(s, o, 32); s2 += __shfl_xor(s2, o, 32); }
  float mean = s / (float)cH;
  float inv  = rsqrtf(s2 / (float)cH - mean * mean + 1e-5f);
#pragma unroll
  for (int j = 0; j < 2; ++j) {
    int c = lane + j * 32;
    float ln = (v[j] - mean) * inv * g[c] + b[c];
    float mv = leakyf(ln);
    int i2 = c & ~1;
    float ang = (float)t * __expf(-(float)i2 * (9.210340371976184f / (float)cH));
    float pe  = (c & 1) ? cosf(ang) : sinf(ang);
    src[row * cH + c] = mv + pe;
  }
}

__global__ __launch_bounds__(256) void k_ln_rows(const float* __restrict__ in,
                                                 float* __restrict__ outp, int rows)
{ // LayerNorm without affine (agg normalization)
  int row = blockIdx.x * 8 + (threadIdx.x >> 5);
  if (row >= rows) return;
  const int lane = threadIdx.x & 31;
  float v[2]; float s = 0.f, s2 = 0.f;
#pragma unroll
  for (int j = 0; j < 2; ++j) {
    int c = lane + j * 32;
    float val = in[(long long)row * cH + c];
    v[j] = val; s += val; s2 += val * val;
  }
  for (int o = 16; o; o >>= 1) { s += __shfl_xor(s, o, 32); s2 += __shfl_xor(s2, o, 32); }
  float mean = s / (float)cH;
  float inv  = rsqrtf(s2 / (float)cH - mean * mean + 1e-5f);
#pragma unroll
  for (int j = 0; j < 2; ++j) {
    int c = lane + j * 32;
    outp[(long long)row * cH + c] = (v[j] - mean) * inv;
  }
}

__global__ void k_extract_rts(const float* __restrict__ g2, float* __restrict__ rts)
{ // res_ts = leaky(g2[:, T-1, :])
  int idx = blockIdx.x * blockDim.x + threadIdx.x;
  if (idx >= cN * cH) return;
  int n = idx >> 6, c = idx & 63;
  rts[idx] = leakyf(g2[((long long)n * cT + (cT - 1)) * cH + c]);
}

__global__ void k_zg_combine(const float* __restrict__ zgin,
                             const float* __restrict__ aggn,
                             const float* __restrict__ rts,
                             float* __restrict__ tmp)
{
  int idx = blockIdx.x * blockDim.x + threadIdx.x;
  if (idx >= cN * cH) return;
  float z = sigmf(zgin[idx]);
  tmp[idx] = z * aggn[idx] + (1.f - z) * rts[idx];
}

__global__ __launch_bounds__(256) void k_final(const float* __restrict__ fus,
    const float* __restrict__ lw, const float* __restrict__ lb,
    float* __restrict__ outp)
{
  int n = blockIdx.x * 8 + (threadIdx.x >> 5);
  if (n >= cN) return;
  const int lane = threadIdx.x & 31;
  float s = 0.f;
#pragma unroll
  for (int j = 0; j < 2; ++j) {
    int c = lane + j * 32;
    s += leakyf(fus[(long long)n * cH + c]) * lw[c];
  }
  for (int o = 16; o; o >>= 1) s += __shfl_xor(s, o, 32);
  if (lane == 0) outp[n] = s + lb[0];
}

// ---------------------------------------------------------------------------
// Host side
// ---------------------------------------------------------------------------
static inline void launch_gemm(hipStream_t st,
    const float* A, int lda, long long sA1, long long sA2,
    const float* B, int ldb, long long sB1, long long sB2,
    const float* bias, float* C, int ldc, long long sC1, long long sC2,
    int M, int Nc, int K, int B1, int nb2, float alpha, int transB, int accum)
{
  dim3 grid((Nc + 63) / 64, (M + 31) / 32, B1 * nb2);
  k_gemm<<<grid, dim3(256), 0, st>>>(A, lda, sA1, sA2, B, ldb, sB1, sB2,
                                     bias, C, ldc, sC1, sC2,
                                     M, Nc, K, nb2, alpha, transB, accum);
}

extern "C" void kernel_launch(void* const* d_in, const int* in_sizes, int n_in,
                              void* d_out, int out_size, void* d_ws, size_t ws_size,
                              hipStream_t stream)
{
  (void)in_sizes; (void)n_in; (void)out_size; (void)ws_size;
  const long long NT = (long long)cN * cT;   // 30720

  const float* x        = (const float*)d_in[0];
  const int*   edge     = (const int*)  d_in[1];
  const float* fc_in_w  = (const float*)d_in[2];
  const float* fc_in_b  = (const float*)d_in[3];
  const float* gwih0    = (const float*)d_in[4];
  const float* gwhh0    = (const float*)d_in[5];
  const float* gbih0    = (const float*)d_in[6];
  const float* gbhh0    = (const float*)d_in[7];
  const float* gwih1    = (const float*)d_in[8];
  const float* gwhh1    = (const float*)d_in[9];
  const float* gbih1    = (const float*)d_in[10];
  const float* gbhh1    = (const float*)d_in[11];
  const float* dm_qw    = (const float*)d_in[12];
  const float* dm_kw    = (const float*)d_in[13];
  const float* dm_vw    = (const float*)d_in[14];
  const float* dm_vb    = (const float*)d_in[15];
  const float* ssa_qw   = (const float*)d_in[16];
  const float* ssa_kw   = (const float*)d_in[17];
  const float* ssa_vw   = (const float*)d_in[18];
  const float* ssa_vb   = (const float*)d_in[19];
  const float* ln1_g    = (const float*)d_in[20];
  const float* ln1_b    = (const float*)d_in[21];
  const float* ln2_g    = (const float*)d_in[22];
  const float* ln2_b    = (const float*)d_in[23];
  const float* mlp_w1   = (const float*)d_in[24];
  const float* mlp_b1   = (const float*)d_in[25];
  const float* mlp_w2   = (const float*)d_in[26];
  const float* mlp_b2   = (const float*)d_in[27];
  const float* mha_in_w = (const float*)d_in[28];
  const float* mha_in_b = (const float*)d_in[29];
  const float* mha_out_w= (const float*)d_in[30];
  const float* mha_out_b= (const float*)d_in[31];
  const float* gf_xs_w  = (const float*)d_in[32];
  const float* gf_xt_w  = (const float*)d_in[33];
  const float* gf_xt_b  = (const float*)d_in[34];
  const float* gf_h_w   = (const float*)d_in[35];
  const float* gf_h_b   = (const float*)d_in[36];
  const float* lin_w    = (const float*)d_in[37];
  const float* lin_b    = (const float*)d_in[38];

  float* out  = (float*)d_out;
  float* attn = out + cN;                 // attn_s region: (T,NH,N,N) fp32

  // ---- workspace bump allocator (deterministic) ----
  char* ws = (char*)d_ws; size_t cur = 0;
  auto alloc = [&](size_t nbytes) -> void* {
    void* p = (void*)(ws + cur);
    cur += (nbytes + 255) & ~(size_t)255;
    return p;
  };
  float* bufH    = (float*)alloc(NT * cH  * 4);
  float* bufGI   = (float*)alloc(NT * 3*cH* 4);   // reused later as MQKV
  float* bufG1   = (float*)alloc(NT * cH  * 4);
  float* bufG2   = (float*)alloc(NT * cH  * 4);
  float* bufXS   = (float*)alloc(NT * cH  * 4);
  float* bufQ    = (float*)alloc(NT * cH  * 4);
  float* bufK    = (float*)alloc(NT * cH  * 4);
  float* bufV    = (float*)alloc(NT * cH  * 4);
  float* bufQS   = (float*)alloc(NT * cH  * 4);
  float* bufKS   = (float*)alloc(NT * cH  * 4);
  float* bufVS   = (float*)alloc(NT * cH  * 4);
  float* bufDYN  = (float*)alloc(NT * cH  * 4);   // (T,NH,N,HD)
  float* bufA2   = (float*)alloc((size_t)cT * cN * cN * 4);
  float* bufSTAT = (float*)alloc(NT * cH  * 4);
  float* bufRES  = (float*)alloc(NT * cH  * 4);
  float* bufO1   = (float*)alloc(NT * cH  * 4);
  float* bufMLP  = (float*)alloc(NT * 2*cH* 4);
  float* bufO2   = (float*)alloc(NT * cH  * 4);
  float* bufSRC  = (float*)alloc(NT * cH  * 4);
  float* bufSC   = (float*)alloc((size_t)cN * cT * cT * 4);
  float* bufMO   = (float*)alloc(NT * cH  * 4);
  float* bufAGG  = (float*)alloc((size_t)cN * cH * 4);
  float* bufAGGN = (float*)alloc((size_t)cN * cH * 4);
  float* bufRTS  = (float*)alloc((size_t)cN * cH * 4);
  float* bufZG   = (float*)alloc((size_t)cN * cH * 4);
  float* bufTMP  = (float*)alloc((size_t)cN * cH * 4);
  float* bufFUS  = (float*)alloc((size_t)cN * cH * 4);
  unsigned char* adjm = (unsigned char*)alloc((size_t)cN * cN);
  float* bufMQKV = bufGI;                 // GI dead after both GRU layers

  const long long NH_ = cNH, NN = cN, HH = cH, HD = cHD, TT = cT;

  // 1. h = x @ fc_in_w^T + b           (K=6 via guarded WMMA fragments)
  launch_gemm(stream, x, cDF,0,0, fc_in_w, cDF,0,0, fc_in_b,
              bufH, cH,0,0, (int)NT, cH, cDF, 1,1, 1.f, 1, 0);
  // 2-3. GRU layer 0
  launch_gemm(stream, bufH, cH,0,0, gwih0, cH,0,0, gbih0,
              bufGI, 3*cH,0,0, (int)NT, 3*cH, cH, 1,1, 1.f, 1, 0);
  k_gru<<<dim3(cN/64), dim3(256), 96*1024, stream>>>(bufGI, gwhh0, gbhh0, bufG1);
  // 4-5. GRU layer 1
  launch_gemm(stream, bufG1, cH,0,0, gwih1, cH,0,0, gbih1,
              bufGI, 3*cH,0,0, (int)NT, 3*cH, cH, 1,1, 1.f, 1, 0);
  k_gru<<<dim3(cN/64), dim3(256), 96*1024, stream>>>(bufGI, gwhh1, gbhh1, bufG2);
  k_extract_rts<<<dim3((cN*cH)/256), 256, 0, stream>>>(bufG2, bufRTS);
  // 6. xs = swapaxes(h,0,1)
  { long long tot = NT * cH;
    k_transpose<<<dim3((unsigned)((tot + 255)/256)), 256, 0, stream>>>(bufH, bufXS); }
  // 7. projections (all (30720,64)x(64,64))
  launch_gemm(stream, bufXS,cH,0,0, dm_qw, cH,0,0, nullptr, bufQ, cH,0,0, (int)NT,cH,cH,1,1,1.f,1,0);
  launch_gemm(stream, bufXS,cH,0,0, dm_kw, cH,0,0, nullptr, bufK, cH,0,0, (int)NT,cH,cH,1,1,1.f,1,0);
  launch_gemm(stream, bufXS,cH,0,0, dm_vw, cH,0,0, dm_vb,  bufV, cH,0,0, (int)NT,cH,cH,1,1,1.f,1,0);
  launch_gemm(stream, bufXS,cH,0,0, ssa_qw,cH,0,0, nullptr, bufQS,cH,0,0, (int)NT,cH,cH,1,1,1.f,1,0);
  launch_gemm(stream, bufXS,cH,0,0, ssa_kw,cH,0,0, nullptr, bufKS,cH,0,0, (int)NT,cH,cH,1,1,1.f,1,0);
  launch_gemm(stream, bufXS,cH,0,0, ssa_vw,cH,0,0, ssa_vb, bufVS,cH,0,0, (int)NT,cH,cH,1,1,1.f,1,0);
  // 8. dynamic-attention scores -> directly into output region
  launch_gemm(stream, bufQ, cH, NN*HH, HD,  bufK, cH, NN*HH, HD, nullptr,
              attn, cN, NH_*NN*NN, NN*NN, cN, cN, cHD, cT, cNH, 0.125f, 1, 0);
  // 9. mean-mask softmax in place on output
  { long long rows = TT * NH_ * NN;
    k_softmax_meanmask<<<dim3((unsigned)((rows + 7)/8)), 256, 0, stream>>>(attn, rows, cN); }
  // 10. dyn = P @ V    (K=512, B non-transposed)
  launch_gemm(stream, attn, cN, NH_*NN*NN, NN*NN, bufV, cH, NN*HH, HD, nullptr,
              bufDYN, cHD, NH_*NN*HD, NN*HD, cN, cHD, cN, cT, cNH, 1.f, 0, 0);
  // 11. static scores a2 = qs @ ks^T / 8
  launch_gemm(stream, bufQS, cH, NN*HH, 0, bufKS, cH, NN*HH, 0, nullptr,
              bufA2, cN, NN*NN, 0, cN, cN, cH, cT, 1, 0.125f, 1, 0);
  // 12. adjacency mask
  k_zero8<<<dim3((cN*cN + 255)/256), 256, 0, stream>>>(adjm, (long long)cN*cN);
  k_adj_scatter<<<dim3((cE + 255)/256), 256, 0, stream>>>(edge, adjm);
  // 13. graph-masked softmax
  { long long rows = TT * NN;
    k_softmax_adjmask<<<dim3((unsigned)((rows + 7)/8)), 256, 0, stream>>>(bufA2, adjm, rows); }
  // 14. stat = a2s @ vs   (K=512, B non-transposed)
  launch_gemm(stream, bufA2, cN, NN*NN, 0, bufVS, cH, NN*HH, 0, nullptr,
              bufSTAT, cH, NN*HH, 0, cN, cH, cN, cT, 1, 1.f, 0, 0);
  // 15. residual = dyn + stat + xs ; o = LN1(residual)
  k_res_ln1<<<dim3((unsigned)((NT + 7)/8)), 256, 0, stream>>>(
      bufDYN, bufSTAT, bufXS, ln1_g, ln1_b, bufRES, bufO1, NT);
  // 16-17. MLP (64->128->64, no activation per reference)
  launch_gemm(stream, bufO1, cH,0,0, mlp_w1, cH,0,0, mlp_b1,
              bufMLP, 2*cH,0,0, (int)NT, 2*cH, cH, 1,1, 1.f, 1, 0);
  launch_gemm(stream, bufMLP, 2*cH,0,0, mlp_w2, 2*cH,0,0, mlp_b2,
              bufO2, cH,0,0, (int)NT, cH, 2*cH, 1,1, 1.f, 1, 0);
  // 18. LN2(o+residual) -> leaky -> + positional encoding
  k_ln2_pe<<<dim3((unsigned)((NT + 7)/8)), 256, 0, stream>>>(
      bufO2, bufRES, ln2_g, ln2_b, bufSRC, NT);
  // 19. mq|mk|mv = src @ mha_in_w^T + b
  launch_gemm(stream, bufSRC, cH,0,0, mha_in_w, cH,0,0, mha_in_b,
              bufMQKV, 3*cH,0,0, (int)NT, 3*cH, cH, 1,1, 1.f, 1, 0);
  // 20. sc[n] = mq_n @ mk_n^T / 8  (batched over nodes, strided rows)
  launch_gemm(stream, bufMQKV, cN*3*cH, 3*cH, 0, bufMQKV + cH, cN*3*cH, 3*cH, 0, nullptr,
              bufSC, cT, TT*TT, 0, cT, cT, cH, cN, 1, 0.125f, 1, 0);
  // 21. softmax over time (len=60)
  { long long rows = NN * TT;
    k_softmax<<<dim3((unsigned)((rows + 7)/8)), 256, 0, stream>>>(bufSC, rows, cT); }
  // 22. moPre[t,n] = probs_n @ mv_n  (K=60 zero-padded in fragments)
  launch_gemm(stream, bufSC, cT, TT*TT, 0, bufMQKV + 2*cH, cN*3*cH, 3*cH, 0, nullptr,
              bufMO, cN*cH, HH, 0, cT, cH, cT, cN, 1, 1.f, 0, 0);
  // 23. agg = moPre[T-1] @ mha_out_w^T + b   (only last timestep needed)
  launch_gemm(stream, bufMO + (long long)(cT-1)*cN*cH, cH,0,0, mha_out_w, cH,0,0,
              mha_out_b, bufAGG, cH,0,0, cN, cH, cH, 1,1, 1.f, 1, 0);
  // 24. normalize agg
  k_ln_rows<<<dim3((cN + 7)/8), 256, 0, stream>>>(bufAGG, bufAGGN, cN);
  // 25. zg = sigmoid(aggn@gf_xs_w^T + rts@gf_xt_w^T + gf_xt_b); gate
  launch_gemm(stream, bufAGGN, cH,0,0, gf_xs_w, cH,0,0, gf_xt_b,
              bufZG, cH,0,0, cN, cH, cH, 1,1, 1.f, 1, 0);
  launch_gemm(stream, bufRTS, cH,0,0, gf_xt_w, cH,0,0, nullptr,
              bufZG, cH,0,0, cN, cH, cH, 1,1, 1.f, 1, 1);   // accumulate
  k_zg_combine<<<dim3((cN*cH)/256), 256, 0, stream>>>(bufZG, bufAGGN, bufRTS, bufTMP);
  // 26. fused pre-activation
  launch_gemm(stream, bufTMP, cH,0,0, gf_h_w, cH,0,0, gf_h_b,
              bufFUS, cH,0,0, cN, cH, cH, 1,1, 1.f, 1, 0);
  // 27. out[n] = leaky(fused) . lin_w + lin_b
  k_final<<<dim3((cN + 7)/8), 256, 0, stream>>>(bufFUS, lin_w, lin_b, out);
}
